// CellLSTM_42846593745221
// MI455X (gfx1250) — matrix operand order repaired and checked by reading, not compiled, not run on hardware
//
#include <hip/hip_runtime.h>

// ---------------------------------------------------------------------------
// Fused LSTM cell for MI455X (gfx1250, wave32, WMMA).
//   z = [x|h] @ [Wi|Wf|Wg|Wo] + b ;  i,f,o = sigmoid, g = tanh
//   c_new = f*c + i*g ; h_new = o*tanh(c_new)
// GEMM on v_wmma_f32_16x16x32_bf16 (fp32 -> bf16 in LDS, f32 accumulate).
// One workgroup accumulates all four gates for a 128x32 output tile (gate
// pre-activations never touch HBM). LDS is double-buffered so global->LDS
// staging for K-step k+1 overlaps the 8 WMMAs of step k.
// ---------------------------------------------------------------------------

typedef __attribute__((ext_vector_type(16))) __bf16 v16bf;
typedef __attribute__((ext_vector_type(8)))  __bf16 v8bf;
typedef __attribute__((ext_vector_type(8)))  float  v8f;

#define B_SZ 8192
#define T_SZ 1024
#define U_SZ 1024

#define BM 128           // batch rows per workgroup
#define BN 32            // units per workgroup
#define BK 32            // K per WMMA step
#define NSTEPS 64        // (T+U)/BK
#define LDSW 40          // padded LDS row stride in bf16 (80B, 8B/16B aligned)

__device__ __forceinline__ float fast_sigmoid(float z) {
    return 1.0f / (1.0f + __expf(-z));
}

union Frag  { v16bf v; v8bf h[2]; };
union Pack4 { __bf16 b[4]; uint2 u; };

__global__ __launch_bounds__(256)
void lstm_cell_wmma_bf16(const float* __restrict__ x,
                         const float* __restrict__ h,
                         const float* __restrict__ c,
                         const float* __restrict__ Wi,
                         const float* __restrict__ Wf,
                         const float* __restrict__ Wg,
                         const float* __restrict__ Wo,
                         const float* __restrict__ bi,
                         const float* __restrict__ bff,
                         const float* __restrict__ bg,
                         const float* __restrict__ bo,
                         float* __restrict__ h_new,
                         float* __restrict__ c_new)
{
    __shared__ __bf16 Alds[2][BM][LDSW];        // [buf][m][k]    2 x 10240 B
    __shared__ __bf16 Blds[2][4][BN][LDSW];     // [buf][g][n][k] 2 x 10240 B

    const int tid  = threadIdx.x;
    const int lane = tid & 31;
    const int wave = tid >> 5;
    const int wm   = wave & 3;   // wave M tile: rows wm*32 .. wm*32+31
    const int wn   = wave >> 2;  // wave N tile: cols wn*16 .. wn*16+15

    const int bm0 = blockIdx.y * BM;
    const int bn0 = blockIdx.x * BN;

    v8f acc[4][2] = {};          // [gate][M-subtile] 16x16 f32 accumulators

    const int half  = lane >> 4;         // 0 | 1
    const int l16   = lane & 15;
    const int a_row = wm * 32 + l16;     // first A-fragment LDS row
    const int b_row = wn * 16 + l16;     // B-fragment LDS row (column of W)

    // A staging: 128 rows x 32 k-floats = 1024 float4 -> 4 per thread
    const int sArow  = tid >> 3;         // 0..31 (+ j*32)
    const int sAquad = (tid & 7) * 4;    // 0,4,..,28
    // W staging: thread owns (n, k-quartet): 4 coalesced global_load_b32,
    // packed into one ds_store_b64 in the transposed [n][k] layout.
    const int sWn  = tid & 31;           // 0..31
    const int sWk0 = (tid >> 5) * 4;     // 0,4,..,28

    // ---- staging helpers (named kernarg pointers => global_load_*) ------
    auto stageA = [&](int ks, int buf) {
        const float* base = (ks < NSTEPS / 2) ? x : h;
        const int kcol = (ks & (NSTEPS / 2 - 1)) * BK;
        const float* p = base + (size_t)(bm0 + sArow) * (size_t)T_SZ + kcol + sAquad;
        #pragma unroll
        for (int j = 0; j < 4; ++j) {
            const float4 v = *(const float4*)(p + (size_t)j * 32u * T_SZ);
            Pack4 pk;
            pk.b[0] = (__bf16)v.x;  pk.b[1] = (__bf16)v.y;
            pk.b[2] = (__bf16)v.z;  pk.b[3] = (__bf16)v.w;
            *(uint2*)&Alds[buf][sArow + j * 32][sAquad] = pk.u;
        }
    };
    auto stageW = [&](int ks, int buf) {
        const size_t kro = (size_t)(ks * BK + sWk0) * (size_t)U_SZ + bn0 + sWn;
        const float* p0 = Wi + kro;
        const float* p1 = Wf + kro;
        const float* p2 = Wg + kro;
        const float* p3 = Wo + kro;
        Pack4 pk;
        pk.b[0] = (__bf16)p0[0];          pk.b[1] = (__bf16)p0[U_SZ];
        pk.b[2] = (__bf16)p0[2 * U_SZ];   pk.b[3] = (__bf16)p0[3 * U_SZ];
        *(uint2*)&Blds[buf][0][sWn][sWk0] = pk.u;
        pk.b[0] = (__bf16)p1[0];          pk.b[1] = (__bf16)p1[U_SZ];
        pk.b[2] = (__bf16)p1[2 * U_SZ];   pk.b[3] = (__bf16)p1[3 * U_SZ];
        *(uint2*)&Blds[buf][1][sWn][sWk0] = pk.u;
        pk.b[0] = (__bf16)p2[0];          pk.b[1] = (__bf16)p2[U_SZ];
        pk.b[2] = (__bf16)p2[2 * U_SZ];   pk.b[3] = (__bf16)p2[3 * U_SZ];
        *(uint2*)&Blds[buf][2][sWn][sWk0] = pk.u;
        pk.b[0] = (__bf16)p3[0];          pk.b[1] = (__bf16)p3[U_SZ];
        pk.b[2] = (__bf16)p3[2 * U_SZ];   pk.b[3] = (__bf16)p3[3 * U_SZ];
        *(uint2*)&Blds[buf][3][sWn][sWk0] = pk.u;
    };

    // ---- software pipeline: stage(k+1) overlaps compute(k) --------------
    stageA(0, 0);
    stageW(0, 0);
    __syncthreads();

    for (int ks = 0; ks < NSTEPS; ++ks) {
        const int cur = ks & 1;
        const int nxt = cur ^ 1;

        if (ks + 1 < NSTEPS) {
            stageA(ks + 1, nxt);
            stageW(ks + 1, nxt);
        }
        if (ks + 2 < NSTEPS) {   // warm L2 two steps ahead (global_prefetch_b8)
            const float* pbase = (ks + 2 < NSTEPS / 2) ? x : h;
            const int kcol2 = ((ks + 2) & (NSTEPS / 2 - 1)) * BK;
            __builtin_prefetch(pbase + (size_t)(bm0 + sArow) * T_SZ + kcol2 + sAquad, 0, 1);
            __builtin_prefetch(Wi + (size_t)((ks + 2) * BK + sWk0) * U_SZ + bn0 + sWn, 0, 1);
        }

        // ---- A fragments (ISA 16-bit A 16x32 layout) -------------------
        // lane<16: K 0-7 (v0-3), 16-23 (v4-7); lane>=16: K 8-15, 24-31
        Frag af0, af1;
        af0.h[0] = *(const v8bf*)&Alds[cur][a_row][half * 8];
        af0.h[1] = *(const v8bf*)&Alds[cur][a_row][16 + half * 8];
        af1.h[0] = *(const v8bf*)&Alds[cur][a_row + 16][half * 8];
        af1.h[1] = *(const v8bf*)&Alds[cur][a_row + 16][16 + half * 8];

        // ---- B fragment per gate, reused for both M sub-tiles ----------
        // B 32x16 layout: lanes 0-15 hold K 0-7, 16-23; lanes 16-31 K 8-15, 24-31
        #pragma unroll
        for (int g = 0; g < 4; ++g) {
            Frag bf;
            bf.h[0] = *(const v8bf*)&Blds[cur][g][b_row][half * 8];
            bf.h[1] = *(const v8bf*)&Blds[cur][g][b_row][16 + half * 8];
            acc[g][0] = __builtin_amdgcn_wmma_f32_16x16x32_bf16(
                            false, af0.v, false, bf.v,
                            (short)0, acc[g][0], false, false);
            acc[g][1] = __builtin_amdgcn_wmma_f32_16x16x32_bf16(
                            false, af1.v, false, bf.v,
                            (short)0, acc[g][1], false, false);
        }

        __syncthreads();
    }

    // ---- epilogue: biases, activations, cell update ---------------------
    // C/D layout: VGPR r, lanes 0-15 -> (M=r, N=lane); lanes 16-31 -> M=r+8
    const int ucol = bn0 + wn * 16 + l16;
    const float vbi = bi[ucol];
    const float vbf = bff[ucol];
    const float vbg = bg[ucol];
    const float vbo = bo[ucol];

    #pragma unroll
    for (int s = 0; s < 2; ++s) {
        #pragma unroll
        for (int r = 0; r < 8; ++r) {
            const int m = half ? (r + 8) : r;
            const size_t row = (size_t)(bm0 + wm * 32 + s * 16 + m);
            const size_t idx = row * (size_t)U_SZ + ucol;

            const float ig = fast_sigmoid(acc[0][s][r] + vbi);
            const float fg = fast_sigmoid(acc[1][s][r] + vbf);
            const float gg = tanhf(acc[2][s][r] + vbg);
            const float og = fast_sigmoid(acc[3][s][r] + vbo);

            const float cold = c[idx];
            const float cn   = fg * cold + ig * gg;
            const float hn   = og * tanhf(cn);

            h_new[idx] = hn;
            c_new[idx] = cn;
        }
    }
}

extern "C" void kernel_launch(void* const* d_in, const int* in_sizes, int n_in,
                              void* d_out, int out_size, void* d_ws, size_t ws_size,
                              hipStream_t stream) {
    (void)in_sizes; (void)n_in; (void)d_ws; (void)ws_size; (void)out_size;

    const float* x  = (const float*)d_in[0];
    const float* h  = (const float*)d_in[1];
    const float* c  = (const float*)d_in[2];
    const float* Wi = (const float*)d_in[3];
    const float* Wf = (const float*)d_in[4];
    const float* Wg = (const float*)d_in[5];
    const float* Wo = (const float*)d_in[6];
    const float* bi = (const float*)d_in[7];
    const float* bf = (const float*)d_in[8];
    const float* bg = (const float*)d_in[9];
    const float* bo = (const float*)d_in[10];

    float* out   = (float*)d_out;
    float* h_new = out;                                   // [B, U]
    float* c_new = out + (size_t)B_SZ * (size_t)U_SZ;     // [B, U]

    dim3 grid(U_SZ / BN, B_SZ / BM);   // (32, 64)
    dim3 block(256);
    lstm_cell_wmma_bf16<<<grid, block, 0, stream>>>(
        x, h, c, Wi, Wf, Wg, Wo, bi, bf, bg, bo, h_new, c_new);
}
